// Block_43593918055013
// MI455X (gfx1250) — matrix operand order repaired and checked by reading, not compiled, not run on hardware
//
#include <hip/hip_runtime.h>
#include <hip/hip_bf16.h>
#include <math.h>

// ---------------------------------------------------------------------------
// CDNA5 (gfx1250) transformer block: bf16 WMMA GEMMs + LDS-resident attention
// ---------------------------------------------------------------------------

typedef __attribute__((ext_vector_type(16))) __bf16 v16bf;
typedef __attribute__((ext_vector_type(8)))  float  v8f;

union FragB16 {
  v16bf v;
  uint4 q[2];
  unsigned short u[16];
};

__device__ __forceinline__ unsigned short f2bf(float f) {
  unsigned int x = __builtin_bit_cast(unsigned int, f);
  x += 0x7fffu + ((x >> 16) & 1u);           // round-to-nearest-even
  return (unsigned short)(x >> 16);
}

__device__ __forceinline__ v8f wmma_bf16(v16bf a, v16bf b, v8f c) {
  // v_wmma_f32_16x16x32_bf16  D = A(16x32) * B(32x16) + C(16x16 f32)
  return __builtin_amdgcn_wmma_f32_16x16x32_bf16(
      /*neg_a=*/false, a, /*neg_b=*/false, b,
      /*c_mod=*/(short)0, c, /*reuse_a=*/false, /*reuse_b=*/false);
}

// A fragment 16x32 bf16: src points at element (m0, k0) of row-major [.,ld].
// ISA layout: lanes 0-15 -> rows, K 0..7 (v0-3) + 16..23 (v4-7); lanes 16-31
// hold K 8..15 + 24..31.
__device__ __forceinline__ v16bf load_a_frag(const unsigned short* __restrict__ src, int ld) {
  int lane = threadIdx.x & 31;
  int m = lane & 15;
  int kb = (lane >> 4) * 8;
  const unsigned short* p = src + (size_t)m * ld + kb;
  FragB16 f;
  f.q[0] = *(const uint4*)(p);        // K kb..kb+7
  f.q[1] = *(const uint4*)(p + 16);   // K kb+16..kb+23
  return f.v;
}

// B fragment 32x16 bf16 from K-contiguous storage (i.e. W^T[n][k] or K-matrix
// rows): lane n holds column n, K kb..kb+7 and kb+16..kb+23 (kb = 8*half).
__device__ __forceinline__ v16bf load_bT_frag(const unsigned short* __restrict__ srcT, int ld) {
  int lane = threadIdx.x & 31;
  int n = lane & 15;
  int kb = (lane >> 4) * 8;
  const unsigned short* p = srcT + (size_t)n * ld + kb;
  FragB16 f;
  f.q[0] = *(const uint4*)(p);        // K kb..kb+7
  f.q[1] = *(const uint4*)(p + 16);   // K kb+16..kb+23
  return f.v;
}

// B fragment 32x16 bf16 from row-major [K,ld] (strided columns; V matrix).
__device__ __forceinline__ v16bf load_bS_frag(const unsigned short* __restrict__ src, int ld) {
  int lane = threadIdx.x & 31;
  int n = lane & 15;
  int kb = (lane >> 4) * 8;
  const unsigned short* p = src + (size_t)kb * ld + n;
  FragB16 f;
#pragma unroll
  for (int j = 0; j < 16; ++j) { f.u[j] = *p; p += (j == 7 ? 9 : 1) * ld; }
  return f.v;
}

// A fragment 16x32 from f32 LDS (softmax probabilities), convert to bf16.
__device__ __forceinline__ v16bf load_aS_frag(const float* __restrict__ S, int sstr,
                                              int m0, int k0) {
  int lane = threadIdx.x & 31;
  const float* r = S + (size_t)(m0 + (lane & 15)) * sstr + k0 + (lane >> 4) * 8;
  FragB16 f;
#pragma unroll
  for (int j = 0; j < 8; ++j) f.u[j] = f2bf(r[j]);
#pragma unroll
  for (int j = 0; j < 8; ++j) f.u[8 + j] = f2bf(r[16 + j]);
  return f.v;
}

// ---------------------------------------------------------------------------
// Weight convert + transpose: w[K,N] f32 -> wt[N,K] bf16
// ---------------------------------------------------------------------------
__global__ void wtrans_kernel(const float* __restrict__ w, unsigned short* __restrict__ wt,
                              int K, int N) {
  size_t i = (size_t)blockIdx.x * blockDim.x + threadIdx.x;
  if (i >= (size_t)K * N) return;
  int n = (int)(i / K);
  int k = (int)(i % K);
  wt[i] = f2bf(w[(size_t)k * N + n]);
}

// ---------------------------------------------------------------------------
// LayerNorm over C=768, output bf16.  One block per row, 256 threads.
// ---------------------------------------------------------------------------
__global__ void __launch_bounds__(256) ln_kernel(const float* __restrict__ x,
                                                 const float* __restrict__ g,
                                                 const float* __restrict__ bta,
                                                 unsigned short* __restrict__ out) {
  constexpr int C = 768;
  __shared__ float s1[256], s2[256];
  int row = blockIdx.x;
  int tid = threadIdx.x;
  const float* xr = x + (size_t)row * C;
  float a = xr[tid], b = xr[tid + 256], c = xr[tid + 512];
  s1[tid] = a + b + c;
  s2[tid] = a * a + b * b + c * c;
  __syncthreads();
  for (int s = 128; s > 0; s >>= 1) {
    if (tid < s) { s1[tid] += s1[tid + s]; s2[tid] += s2[tid + s]; }
    __syncthreads();
  }
  float mu = s1[0] * (1.0f / C);
  float var = s2[0] * (1.0f / C) - mu * mu;
  float rstd = rsqrtf(var + 1e-5f);
  unsigned short* orow = out + (size_t)row * C;
  orow[tid]       = f2bf((a - mu) * rstd * g[tid]       + bta[tid]);
  orow[tid + 256] = f2bf((b - mu) * rstd * g[tid + 256] + bta[tid + 256]);
  orow[tid + 512] = f2bf((c - mu) * rstd * g[tid + 512] + bta[tid + 512]);
}

// ---------------------------------------------------------------------------
// GEMM: out[M,N] = A[M,K](bf16) @ W[K,N] + bias, W stored transposed bf16.
// Block tile 128x128, 8 waves; wave tile 32x64 (2x4 WMMA accumulators).
// MODE 0: bf16 out.  MODE 1: f32 out = resid + v.  MODE 2: gelu(v) -> bf16.
// ---------------------------------------------------------------------------
template <int MODE>
__global__ void __launch_bounds__(256) gemm_wmma(const unsigned short* __restrict__ A,
                                                 const unsigned short* __restrict__ WT,
                                                 const float* __restrict__ bias,
                                                 const float* __restrict__ resid,
                                                 float* __restrict__ outF,
                                                 unsigned short* __restrict__ outB,
                                                 int M, int N, int K) {
  int w = threadIdx.x >> 5;
  int lane = threadIdx.x & 31;
  int m0 = blockIdx.y * 128 + (w & 3) * 32;
  int n0 = blockIdx.x * 128 + (w >> 2) * 64;

  v8f acc[2][4] = {};
  for (int k0 = 0; k0 < K; k0 += 32) {
    v16bf a0 = load_a_frag(A + (size_t)m0 * K + k0, K);
    v16bf a1 = load_a_frag(A + (size_t)(m0 + 16) * K + k0, K);
#pragma unroll
    for (int t = 0; t < 4; ++t) {
      v16bf b = load_bT_frag(WT + (size_t)(n0 + 16 * t) * K + k0, K);
      acc[0][t] = wmma_bf16(a0, b, acc[0][t]);
      acc[1][t] = wmma_bf16(a1, b, acc[1][t]);
    }
  }

  int half = lane >> 4;
  int nc = lane & 15;
#pragma unroll
  for (int i = 0; i < 2; ++i) {
#pragma unroll
    for (int t = 0; t < 4; ++t) {
      int col = n0 + 16 * t + nc;
      float bv = bias[col];
#pragma unroll
      for (int r = 0; r < 8; ++r) {
        int row = m0 + 16 * i + r + half * 8;
        float v = acc[i][t][r] + bv;
        size_t idx = (size_t)row * N + col;
        if (MODE == 0) {
          outB[idx] = f2bf(v);
        } else if (MODE == 1) {
          outF[idx] = resid[idx] + v;
        } else {
          float gl = 0.5f * v * (1.0f + erff(v * 0.70710678118f));
          outB[idx] = f2bf(gl);
        }
      }
    }
  }
}

// ---------------------------------------------------------------------------
// Attention: per block = (b, h, 32 query rows).  S[32x1024] f32 in LDS.
//   phase 1: S = Q K^T (WMMA), phase 2: scaled softmax in LDS,
//   phase 3: O = P V (WMMA), bf16 out at [row, h*64+d].
// qkv layout: row (b*1024+n), columns [i3*768 + h*64 + d], bf16.
// ---------------------------------------------------------------------------
__global__ void __launch_bounds__(256) attn_wmma(const unsigned short* __restrict__ qkv,
                                                 unsigned short* __restrict__ o) {
  constexpr int Ns = 1024, C3 = 2304, Cc = 768;
  constexpr int SSTR = 1028;                 // padded row stride (bank spread)
  extern __shared__ float smem[];
  float* S = smem;                           // 32 x SSTR
  float* red = smem + 32 * SSTR;             // 256 reduction slots

  int qb = blockIdx.x & 31;
  int hh = (blockIdx.x >> 5) % 12;
  int bb = blockIdx.x / (32 * 12);

  int tid = threadIdx.x;
  int w = tid >> 5;
  int lane = tid & 31;
  int half = lane >> 4;
  int nc = lane & 15;

  size_t qrow0 = (size_t)bb * Ns + (size_t)qb * 32;

  // ---- Phase 1: S = Q K^T ----
  {
    int m0 = (w & 1) * 16;                   // 2 row halves
    int c0 = (w >> 1) * 256;                 // 4 key segments of 256
    v16bf aq[2];
#pragma unroll
    for (int ks = 0; ks < 2; ++ks)
      aq[ks] = load_a_frag(qkv + (qrow0 + m0) * C3 + hh * 64 + ks * 32, C3);
#pragma unroll
    for (int nb = 0; nb < 2; ++nb) {
      int n0 = c0 + nb * 128;
      v8f acc[8] = {};
#pragma unroll
      for (int ks = 0; ks < 2; ++ks) {
#pragma unroll
        for (int t = 0; t < 8; ++t) {
          v16bf bk = load_bT_frag(
              qkv + ((size_t)bb * Ns + n0 + 16 * t) * C3 + Cc + hh * 64 + ks * 32, C3);
          acc[t] = wmma_bf16(aq[ks], bk, acc[t]);
        }
      }
#pragma unroll
      for (int t = 0; t < 8; ++t)
#pragma unroll
        for (int r = 0; r < 8; ++r)
          S[(size_t)(m0 + r + half * 8) * SSTR + n0 + 16 * t + nc] = acc[t][r];
    }
  }
  __syncthreads();

  // ---- Phase 2: softmax rows (scale = Hd^-0.5 = 0.125) ----
  {
    int row = tid >> 3;
    int seg = tid & 7;
    float* Sr = S + (size_t)row * SSTR + seg * 128;
    const float scale = 0.125f;
    float mx = -3.0e38f;
    for (int j = 0; j < 128; ++j) mx = fmaxf(mx, Sr[j] * scale);
    red[row * 8 + seg] = mx;
    __syncthreads();
    float rmx = red[row * 8];
#pragma unroll
    for (int j = 1; j < 8; ++j) rmx = fmaxf(rmx, red[row * 8 + j]);
    __syncthreads();
    float sum = 0.f;
    for (int j = 0; j < 128; ++j) {
      float e = __expf(Sr[j] * scale - rmx);
      Sr[j] = e;
      sum += e;
    }
    red[row * 8 + seg] = sum;
    __syncthreads();
    float rs = 0.f;
#pragma unroll
    for (int j = 0; j < 8; ++j) rs += red[row * 8 + j];
    float inv = 1.0f / rs;
    for (int j = 0; j < 128; ++j) Sr[j] *= inv;
  }
  __syncthreads();

  // ---- Phase 3: O = P V ----
  {
    int m0 = (w & 1) * 16;
    int d0 = (w >> 1) * 16;                  // 4 head-dim tiles of 16
    v8f acc = {};
    for (int k0 = 0; k0 < Ns; k0 += 32) {
      v16bf a = load_aS_frag(S, SSTR, m0, k0);
      v16bf bv = load_bS_frag(
          qkv + ((size_t)bb * Ns + k0) * C3 + 2 * Cc + hh * 64 + d0, C3);
      acc = wmma_bf16(a, bv, acc);
    }
    int col = hh * 64 + d0 + nc;
#pragma unroll
    for (int r = 0; r < 8; ++r) {
      size_t row = qrow0 + m0 + r + half * 8;
      o[row * Cc + col] = f2bf(acc[r]);
    }
  }
}

// ---------------------------------------------------------------------------
// Host-side orchestration
// ---------------------------------------------------------------------------
static inline int cdiv(long long a, long long b) { return (int)((a + b - 1) / b); }

extern "C" void kernel_launch(void* const* d_in, const int* in_sizes, int n_in,
                              void* d_out, int out_size, void* d_ws, size_t ws_size,
                              hipStream_t stream) {
  (void)in_sizes; (void)n_in; (void)out_size; (void)ws_size;
  const float* x     = (const float*)d_in[0];
  const float* ln1g  = (const float*)d_in[1];
  const float* ln1b  = (const float*)d_in[2];
  const float* qkvw  = (const float*)d_in[3];
  const float* qkvb  = (const float*)d_in[4];
  const float* projw = (const float*)d_in[5];
  const float* projb = (const float*)d_in[6];
  const float* ln2g  = (const float*)d_in[7];
  const float* ln2b  = (const float*)d_in[8];
  const float* fc1w  = (const float*)d_in[9];
  const float* fc1b  = (const float*)d_in[10];
  const float* fc2w  = (const float*)d_in[11];
  const float* fc2b  = (const float*)d_in[12];
  float* out = (float*)d_out;

  constexpr int Bv = 16, Nv = 1024, C = 768, C3 = 2304, Hid = 3072;
  constexpr int M = Bv * Nv;                 // 16384 rows

  char* ws = (char*)d_ws;
  auto alloc = [&](size_t bytes) {
    char* p = ws;
    ws += (bytes + 255) & ~(size_t)255;
    return p;
  };
  unsigned short* wtqkv  = (unsigned short*)alloc((size_t)C3 * C * 2);
  unsigned short* wtproj = (unsigned short*)alloc((size_t)C * C * 2);
  unsigned short* wtfc1  = (unsigned short*)alloc((size_t)Hid * C * 2);
  unsigned short* wtfc2  = (unsigned short*)alloc((size_t)C * Hid * 2);
  unsigned short* hb     = (unsigned short*)alloc((size_t)M * C * 2);
  unsigned short* qkvo   = (unsigned short*)alloc((size_t)M * C3 * 2);
  unsigned short* ob     = (unsigned short*)alloc((size_t)M * C * 2);
  float*          x2     = (float*)alloc((size_t)M * C * 4);
  unsigned short* h2     = (unsigned short*)alloc((size_t)M * C * 2);
  unsigned short* act    = (unsigned short*)alloc((size_t)M * Hid * 2);

  // 1) weight convert + transpose to bf16 [N,K]
  wtrans_kernel<<<cdiv((long long)C * C3, 256), 256, 0, stream>>>(qkvw, wtqkv, C, C3);
  wtrans_kernel<<<cdiv((long long)C * C,   256), 256, 0, stream>>>(projw, wtproj, C, C);
  wtrans_kernel<<<cdiv((long long)C * Hid, 256), 256, 0, stream>>>(fc1w, wtfc1, C, Hid);
  wtrans_kernel<<<cdiv((long long)Hid * C, 256), 256, 0, stream>>>(fc2w, wtfc2, Hid, C);

  // 2) LN1 -> bf16
  ln_kernel<<<M, 256, 0, stream>>>(x, ln1g, ln1b, hb);

  // 3) qkv = h @ Wqkv + b  (bf16 out)
  gemm_wmma<0><<<dim3(C3 / 128, M / 128), 256, 0, stream>>>(
      hb, wtqkv, qkvb, nullptr, nullptr, qkvo, M, C3, C);

  // 4) attention -> bf16 o
  const size_t attn_smem = (size_t)(32 * 1028 + 256) * sizeof(float);   // ~132 KB
  attn_wmma<<<Bv * 12 * (Nv / 32), 256, attn_smem, stream>>>(qkvo, ob);

  // 5) x2 = x + o @ Wproj + b  (f32)
  gemm_wmma<1><<<dim3(C / 128, M / 128), 256, 0, stream>>>(
      ob, wtproj, projb, x, x2, nullptr, M, C, C);

  // 6) LN2 -> bf16
  ln_kernel<<<M, 256, 0, stream>>>(x2, ln2g, ln2b, h2);

  // 7) act = gelu(h2 @ Wfc1 + b)  (bf16)
  gemm_wmma<2><<<dim3(Hid / 128, M / 128), 256, 0, stream>>>(
      h2, wtfc1, fc1b, nullptr, nullptr, act, M, Hid, C);

  // 8) out = x2 + act @ Wfc2 + b  (f32)
  gemm_wmma<1><<<dim3(C / 128, M / 128), 256, 0, stream>>>(
      act, wtfc2, fc2b, x2, out, nullptr, M, C, Hid);
}
